// PICODETLoss_19988777796201
// MI455X (gfx1250) — matrix operand order repaired and checked by reading, not compiled, not run on hardware
//
#include <hip/hip_runtime.h>
#include <hip/hip_bf16.h>
#include <math.h>

typedef __attribute__((ext_vector_type(16))) _Float16 v16h;
typedef __attribute__((ext_vector_type(8)))  float    v8f;

#define NB   32
#define NPRI 8500
#define NPAD 8512      // 532 * 16
#define NG   128
#define NC   80
#define KPAD 96        // 3 WMMA K-chunks of 32

// ---------------------------------------------------------------------------
// K1: per-prior class prep: neg_sum[n] and Q[n,c] = (-log p) - (-log1p(-p)),
// stored f16 zero-padded to [NPAD][KPAD] for the WMMA GEMM.
// ---------------------------------------------------------------------------
__global__ __launch_bounds__(256)
void k_prep(const float* __restrict__ clsp, _Float16* __restrict__ Q,
            float* __restrict__ negsum, int b)
{
    int n = blockIdx.x * 256 + threadIdx.x;
    if (n >= NPAD) return;
    _Float16* q = Q + (size_t)n * KPAD;
    if (n < NPRI) {
        const float* p = clsp + ((size_t)b * NPRI + n) * NC;
        float ns = 0.f;
        for (int c = 0; c < NC; ++c) {
            float pc  = fminf(fmaxf(p[c], 1e-7f), 1.f - 1e-7f);
            float neg = -log1pf(-pc);
            float pos = -logf(pc);
            ns += neg;
            q[c] = (_Float16)(pos - neg);
        }
        for (int c = NC; c < KPAD; ++c) q[c] = (_Float16)0.0f;
        negsum[n] = ns;
    } else {
        for (int c = 0; c < KPAD; ++c) q[c] = (_Float16)0.0f;
    }
}

// ---------------------------------------------------------------------------
// K2: per-prior geometry vs all GTs: IoU, in_box&in_ctr bitmask, valid flag.
// ---------------------------------------------------------------------------
__global__ __launch_bounds__(256)
void k_geom(const float* __restrict__ priors, const float* __restrict__ dec,
            const float* __restrict__ gtb, float* __restrict__ iou,
            unsigned* __restrict__ msk, unsigned* __restrict__ valid, int b)
{
    __shared__ float4 sgt[NG];
    int tid = threadIdx.x;
    if (tid < NG) sgt[tid] = ((const float4*)(gtb + (size_t)b * NG * 4))[tid];
    __syncthreads();
    int n = blockIdx.x * 256 + tid;
    if (n >= NPRI) return;

    float cx = priors[n*4+0], cy = priors[n*4+1];
    float sw = priors[n*4+2], sh = priors[n*4+3];
    const float* db = dec + ((size_t)b * NPRI + n) * 4;
    float ax0 = db[0], ay0 = db[1], ax1 = db[2], ay1 = db[3];
    float aa = (ax1 - ax0) * (ay1 - ay0);

    unsigned mw[4] = {0u, 0u, 0u, 0u};
    bool anyv = false;
    for (int g = 0; g < NG; ++g) {
        float4 gb = sgt[g];
        float inb = fminf(fminf(cx - gb.x, cy - gb.y), fminf(gb.z - cx, gb.w - cy));
        float gcx = (gb.x + gb.z) * 0.5f, gcy = (gb.y + gb.w) * 0.5f;
        float inc = fminf(fminf(cx - (gcx - 2.5f * sw), cy - (gcy - 2.5f * sh)),
                          fminf(gcx + 2.5f * sw - cx, gcy + 2.5f * sh - cy));
        bool ib = inb > 0.f, ic = inc > 0.f;
        anyv |= (ib | ic);
        if (ib && ic) mw[g >> 5] |= 1u << (g & 31);

        float ab  = (gb.z - gb.x) * (gb.w - gb.y);
        float tlx = fmaxf(ax0, gb.x), tly = fmaxf(ay0, gb.y);
        float brx = fminf(ax1, gb.z), bry = fminf(ay1, gb.w);
        float iw  = fmaxf(brx - tlx, 0.f), ih = fmaxf(bry - tly, 0.f);
        float inter = iw * ih;
        iou[(size_t)n * NG + g] = inter / (aa + ab - inter + 1e-16f);
    }
    msk[n*4+0] = mw[0]; msk[n*4+1] = mw[1]; msk[n*4+2] = mw[2]; msk[n*4+3] = mw[3];
    valid[n] = anyv ? 1u : 0u;
}

// ---------------------------------------------------------------------------
// K3: WMMA one-hot GEMM (Q[N,96] x onehot(labels)[96,128]) fused with the full
// SimOTA cost. One wave32 per 16x16 output tile; 3x v_wmma_f32_16x16x32_f16.
// Register layouts per CDNA5 ISA 7.12.2 (wave32).
// ---------------------------------------------------------------------------
__global__ __launch_bounds__(32)
void k_cost(const _Float16* __restrict__ Q, const float* __restrict__ negsum,
            const float* __restrict__ iou, const unsigned* __restrict__ msk,
            const unsigned* __restrict__ valid, const int* __restrict__ gtl,
            float* __restrict__ cost, int b)
{
    const int lane = threadIdx.x;
    const int m0   = blockIdx.x * 16;
    const int n0   = blockIdx.y * 16;
    const int half = (lane >= 16) ? 1 : 0;
    const int mrow = m0 + (lane & 15);
    const int lbl  = gtl[b * NG + n0 + (lane & 15)];

    const _Float16* qrow = Q + (size_t)mrow * KPAD;
    v8f c = {};
    #pragma unroll
    for (int kc = 0; kc < KPAD; kc += 32) {
        v16h a, bm;
        #pragma unroll
        for (int v = 0; v < 8; ++v) {
            // A (16x32 f16, MxK): vgpr v holds K = 2v (+8 if v>=4) (+8 for hi lanes)
            int kA = kc + 2 * v + (v >= 4 ? 8 : 0) + (half ? 8 : 0);
            a[2*v]   = qrow[kA];
            a[2*v+1] = qrow[kA + 1];
            // B (32x16 f16, KxN): vgpr v holds K = 2v (+8 if v>=4) (+8 for hi lanes), N = lane&15
            int kB = kc + 2 * v + (v >= 4 ? 8 : 0) + (half ? 8 : 0);
            bm[2*v]   = (kB     == lbl) ? (_Float16)1.0f : (_Float16)0.0f;
            bm[2*v+1] = (kB + 1 == lbl) ? (_Float16)1.0f : (_Float16)0.0f;
        }
        c = __builtin_amdgcn_wmma_f32_16x16x32_f16(false, a, false, bm,
                                                   (short)0, c, false, false);
    }

    // C/D layout: vgpr v -> M = v + 8*half, N = lane&15
    const int g = n0 + (lane & 15);
    #pragma unroll
    for (int v = 0; v < 8; ++v) {
        int row = m0 + v + (half ? 8 : 0);
        if (row < NPRI) {
            float clsc = negsum[row] + c[v];                       // CLS_W = 1
            float iv   = iou[(size_t)row * NG + g];
            float iouc = -logf(iv + 1e-8f);                        // IOU_W = 3
            unsigned mw = msk[row * 4 + (g >> 5)];
            bool inb = (mw >> (g & 31)) & 1u;
            bool vd  = valid[row] != 0u;
            cost[(size_t)row * NG + g] =
                clsc + 3.0f * iouc + (inb ? 0.f : 1e5f) + (vd ? 0.f : 1e8f);
        }
    }
}

// ---------------------------------------------------------------------------
// K4: per-GT column: dyn_k = clamp(floor(sum top-10 masked IoU),1,10);
// threshold = dyn_k-th smallest cost. Register top-10 + LDS pair reductions.
// ---------------------------------------------------------------------------
__global__ __launch_bounds__(256)
void k_thresh(const float* __restrict__ cost, const float* __restrict__ iou,
              const unsigned* __restrict__ valid, float* __restrict__ thresh)
{
    __shared__ float ls_c[2560];
    __shared__ float ls_i[2560];
    __shared__ float red[256];
    __shared__ int   redi[256];
    __shared__ float sel[10];
    __shared__ float s_sum;
    __shared__ int   s_dynk;

    const int g = blockIdx.x, tid = threadIdx.x;
    float tc[10], ti[10];
    #pragma unroll
    for (int j = 0; j < 10; ++j) { tc[j] = 3.4e38f; ti[j] = -1.f; }

    for (int n = tid; n < NPRI; n += 256) {
        float cc = cost[(size_t)n * NG + g];
        #pragma unroll
        for (int j = 0; j < 10; ++j)
            if (cc < tc[j]) { float t = tc[j]; tc[j] = cc; cc = t; }
        float vv = valid[n] ? iou[(size_t)n * NG + g] : 0.f;
        #pragma unroll
        for (int j = 0; j < 10; ++j)
            if (vv > ti[j]) { float t = ti[j]; ti[j] = vv; vv = t; }
    }
    #pragma unroll
    for (int j = 0; j < 10; ++j) { ls_c[tid*10+j] = tc[j]; ls_i[tid*10+j] = ti[j]; }
    if (tid == 0) s_sum = 0.f;
    __syncthreads();

    // extract top-10 largest masked IoUs -> dyn_k
    for (int it = 0; it < 10; ++it) {
        float best = -1e30f; int bi = tid * 10;
        #pragma unroll
        for (int j = 0; j < 10; ++j) {
            float v = ls_i[tid*10+j];
            if (v > best) { best = v; bi = tid*10+j; }
        }
        red[tid] = best; redi[tid] = bi; __syncthreads();
        for (int s = 128; s > 0; s >>= 1) {
            if (tid < s && red[tid+s] > red[tid]) { red[tid] = red[tid+s]; redi[tid] = redi[tid+s]; }
            __syncthreads();
        }
        if (tid == 0) { s_sum += fmaxf(red[0], 0.f); ls_i[redi[0]] = -1e30f; }
        __syncthreads();
    }
    if (tid == 0) {
        int k = (int)s_sum;
        s_dynk = k < 1 ? 1 : (k > 10 ? 10 : k);
    }
    __syncthreads();

    // extract 10 smallest costs; threshold = sel[dyn_k-1]
    for (int it = 0; it < 10; ++it) {
        float best = 3.5e38f; int bi = tid * 10;
        #pragma unroll
        for (int j = 0; j < 10; ++j) {
            float v = ls_c[tid*10+j];
            if (v < best) { best = v; bi = tid*10+j; }
        }
        red[tid] = best; redi[tid] = bi; __syncthreads();
        for (int s = 128; s > 0; s >>= 1) {
            if (tid < s && red[tid+s] < red[tid]) { red[tid] = red[tid+s]; redi[tid] = redi[tid+s]; }
            __syncthreads();
        }
        if (tid == 0) { sel[it] = red[0]; ls_c[redi[0]] = 3.5e38f; }
        __syncthreads();
    }
    if (tid == 0) thresh[g] = sel[s_dynk - 1];
}

// ---------------------------------------------------------------------------
// K5: per-prior matching resolution + VFL + GIoU + DFL, batch accumulators.
// ---------------------------------------------------------------------------
__global__ __launch_bounds__(256)
void k_loss(const float* __restrict__ priors, const float* __restrict__ dec,
            const float* __restrict__ clsp, const float* __restrict__ regp,
            const float* __restrict__ gtb, const int* __restrict__ gtl,
            const float* __restrict__ cost, const float* __restrict__ iou,
            const float* __restrict__ thresh, float* __restrict__ acc, int b)
{
    __shared__ float sth[NG];
    __shared__ float sred[256];
    const int tid = threadIdx.x;
    if (tid < NG) sth[tid] = thresh[tid];
    __syncthreads();

    const int n = blockIdx.x * 256 + tid;
    float vfl = 0.f, gis = 0.f, dfs = 0.f, npos = 0.f, nsc = 0.f;
    if (n < NPRI) {
        const float* crow = cost + (size_t)n * NG;
        const float* irow = iou  + (size_t)n * NG;
        int cnt = 0, firstg = 0, bestg = 0;
        float firstiou = 0.f, mincost = 3.5e38f;
        for (int g = 0; g < NG; ++g) {
            float cst = crow[g];
            if (cst < mincost) { mincost = cst; bestg = g; }
            if (cst <= sth[g]) {
                if (cnt == 0) { firstg = g; firstiou = irow[g]; }
                cnt++;
            }
        }
        bool  fg     = cnt > 0;
        int   mgt    = (cnt > 1) ? bestg : firstg;
        float mious  = (cnt > 1) ? irow[bestg] : firstiou;
        float tscore = fg ? mious : 0.f;
        int   lbl    = gtl[b * NG + mgt];

        // Varifocal loss over 80 classes
        const float* p = clsp + ((size_t)b * NPRI + n) * NC;
        for (int c = 0; c < NC; ++c) {
            float pc  = fminf(fmaxf(p[c], 1e-7f), 1.f - 1e-7f);
            float lp  = logf(pc), l1 = log1pf(-pc);
            float tgt = (c == lbl) ? tscore : 0.f;
            float bce = -(tgt * lp + (1.f - tgt) * l1);
            float w   = (tgt > 0.f) ? tgt : 0.75f * pc * pc;   // VFL_ALPHA * p^2
            vfl += bce * w;
        }

        // GIoU loss, weighted by tscore
        const float* db = dec + ((size_t)b * NPRI + n) * 4;
        const float* tb = gtb + ((size_t)b * NG + mgt) * 4;
        float px0 = db[0], py0 = db[1], px1 = db[2], py1 = db[3];
        float tx0 = tb[0], ty0 = tb[1], tx1 = tb[2], ty1 = tb[3];
        float tlx = fmaxf(px0, tx0), tly = fmaxf(py0, ty0);
        float brx = fminf(px1, tx1), bry = fminf(py1, ty1);
        float iw  = fmaxf(brx - tlx, 0.f), ih = fmaxf(bry - tly, 0.f);
        float inter = iw * ih;
        float ap = fmaxf(px1 - px0, 0.f) * fmaxf(py1 - py0, 0.f);
        float at = fmaxf(tx1 - tx0, 0.f) * fmaxf(ty1 - ty0, 0.f);
        float un = ap + at - inter;
        float iv = inter / (un + 1e-16f);
        float cx0 = fminf(px0, tx0), cy0 = fminf(py0, ty0);
        float cx1 = fmaxf(px1, tx1), cy1 = fmaxf(py1, ty1);
        float ac2 = fmaxf(cx1 - cx0, 0.f) * fmaxf(cy1 - cy0, 0.f);
        gis = (1.f - (iv - (ac2 - un) / (ac2 + 1e-16f))) * tscore;

        // DFL over 4 sides x 8 bins
        float cx = priors[n*4+0], cy = priors[n*4+1], s = priors[n*4+2];
        float d4[4] = { (cx - tx0) / s, (cy - ty0) / s, (tx1 - cx) / s, (ty1 - cy) / s };
        const float* rp = regp + ((size_t)b * NPRI + n) * 32;
        float dfl = 0.f;
        for (int j = 0; j < 4; ++j) {
            const float* l = rp + j * 8;
            float m = l[0];
            for (int k = 1; k < 8; ++k) m = fmaxf(m, l[k]);
            float se = 0.f;
            for (int k = 0; k < 8; ++k) se += expf(l[k] - m);
            float lse = logf(se) + m;
            float t  = fminf(fmaxf(d4[j], 0.f), 6.9f);
            int   tl = (int)t;
            int   tr = (tl + 1 > 7) ? 7 : tl + 1;
            float wl = (float)(tl + 1) - t, wr = t - (float)tl;
            dfl += -((l[tl] - lse) * wl + (l[tr] - lse) * wr);
        }
        dfs  = dfl * tscore;
        npos = fg ? 1.f : 0.f;
        nsc  = tscore;
    }

    float vals[5] = { vfl, gis, dfs, npos, nsc };
    for (int i = 0; i < 5; ++i) {
        sred[tid] = vals[i]; __syncthreads();
        for (int s2 = 128; s2 > 0; s2 >>= 1) {
            if (tid < s2) sred[tid] += sred[tid + s2];
            __syncthreads();
        }
        if (tid == 0) atomicAdd(&acc[i], sred[0]);
        __syncthreads();
    }
}

__global__ void k_zero(float* acc) { if (threadIdx.x < 8) acc[threadIdx.x] = 0.f; }

__global__ void k_final(const float* __restrict__ acc, float* __restrict__ out)
{
    float np = fmaxf(acc[3], 1.f), ns = fmaxf(acc[4], 1.f);
    out[0] = acc[0] / np + 2.f * acc[1] / ns + 0.25f * acc[2] / ns;
}

// ---------------------------------------------------------------------------
static inline size_t alup(size_t x) { return (x + 255) & ~(size_t)255; }

extern "C" void kernel_launch(void* const* d_in, const int* in_sizes, int n_in,
                              void* d_out, int out_size, void* d_ws, size_t ws_size,
                              hipStream_t stream)
{
    (void)in_sizes; (void)n_in; (void)out_size; (void)ws_size;
    const float* priors = (const float*)d_in[0];
    const float* dec    = (const float*)d_in[1];
    const float* clsp   = (const float*)d_in[2];
    const float* regp   = (const float*)d_in[3];
    const float* gtb    = (const float*)d_in[4];
    const int*   gtl    = (const int*)d_in[5];

    char* w = (char*)d_ws;
    size_t o = 0;
    _Float16* Q      = (_Float16*)(w + o); o = alup(o + (size_t)NPAD * KPAD * 2);
    float*    negsum = (float*)(w + o);    o = alup(o + (size_t)NPRI * 4);
    float*    iou    = (float*)(w + o);    o = alup(o + (size_t)NPRI * NG * 4);
    float*    cost   = (float*)(w + o);    o = alup(o + (size_t)NPRI * NG * 4);
    unsigned* msk    = (unsigned*)(w + o); o = alup(o + (size_t)NPRI * 16);
    unsigned* valid  = (unsigned*)(w + o); o = alup(o + (size_t)NPRI * 4);
    float*    thresh = (float*)(w + o);    o = alup(o + (size_t)NG * 4);
    float*    acc    = (float*)(w + o);    o = alup(o + 64);

    const int blkN = (NPRI + 255) / 256;   // 34
    const int blkP = (NPAD + 255) / 256;   // 34

    k_zero<<<1, 32, 0, stream>>>(acc);
    for (int b = 0; b < NB; ++b) {
        k_prep<<<blkP, 256, 0, stream>>>(clsp, Q, negsum, b);
        k_geom<<<blkN, 256, 0, stream>>>(priors, dec, gtb, iou, msk, valid, b);
        k_cost<<<dim3(NPAD / 16, NG / 16), 32, 0, stream>>>(Q, negsum, iou, msk,
                                                            valid, gtl, cost, b);
        k_thresh<<<NG, 256, 0, stream>>>(cost, iou, valid, thresh);
        k_loss<<<blkN, 256, 0, stream>>>(priors, dec, clsp, regp, gtb, gtl,
                                         cost, iou, thresh, acc, b);
    }
    k_final<<<1, 1, 0, stream>>>(acc, (float*)d_out);
}
